// KernelLayer_86036784874092
// MI455X (gfx1250) — compile-verified, matrix operand order checked by reading, not run on hardware
//
#include <hip/hip_runtime.h>
#include <stdint.h>

// ---------------------------------------------------------------------------
// RBF kernel layer for MI455X (gfx1250, wave32, WMMA).
//   x  : [4096 x 4096] f32     W : [4096 x 1024] f32
//   out: [4096 x 1024] f32 = exp(-0.5*sqdist(x,x)) @ W
//
// Pipeline (all matrix math via v_wmma_f32_16x16x32_bf16, f32 accum):
//   1) xb  = bf16(x); sq[i] = sum(xb[i]^2)  (norms from ROUNDED data so the
//      diagonal of sqdist cancels exactly -> K diag == 1 in bf16 path)
//   2) Wt  = bf16(W^T)  [1024 x 4096]
//   3) Kb  = exp(-0.5*(sq_i + sq_j - 2 * xb@xb^T))  as bf16 [4096 x 4096]
//   4) out = Kb @ Wt^T  (f32 output)
// GEMMs use 128x128 block tiles with DOUBLE-BUFFERED LDS panels: one
// workgroup barrier per 32-deep k-step; next-tile global loads and LDS
// stores overlap the WMMA stream of the current tile.
// Workspace: xb 32MB @0, Kb 32MB @32MB, Wt 8MB @64MB, sq 16KB @72MB.
// ---------------------------------------------------------------------------

typedef __attribute__((ext_vector_type(16))) __bf16 v16bf;
typedef __attribute__((ext_vector_type(8)))  float  v8f;

constexpr int Dd   = 4096;   // inner dim (and rows of x)
constexpr int OUTC = 1024;   // output columns
constexpr int PK   = 40;     // padded LDS row stride (bf16 elems): 32 + 8 pad

__device__ __forceinline__ uint16_t f32_to_bf16(float f) {
    uint32_t u = __float_as_uint(f);
    u += 0x7FFFu + ((u >> 16) & 1u);     // round-to-nearest-even
    return (uint16_t)(u >> 16);
}
__device__ __forceinline__ float bf16_to_f32(uint16_t h) {
    return __uint_as_float(((uint32_t)h) << 16);
}

union FragBF { uint4 u[2]; v16bf v; };

// -------------------- 1) convert x -> bf16 + row sq-norms -------------------
__global__ __launch_bounds__(256)
void rows_bf16_sqnorm(const float* __restrict__ x,
                      uint16_t* __restrict__ xb,
                      float* __restrict__ sq) {
    __shared__ float red[256];
    const int row = blockIdx.x;
    const int tid = threadIdx.x;
    const float*  xr  = x  + (size_t)row * Dd;
    uint16_t*     xbr = xb + (size_t)row * Dd;
    float s = 0.0f;
    for (int c = tid; c < Dd; c += 256) {
        uint16_t h = f32_to_bf16(xr[c]);
        xbr[c] = h;
        float f = bf16_to_f32(h);
        s += f * f;
    }
    red[tid] = s;
    __syncthreads();
    for (int off = 128; off > 0; off >>= 1) {
        if (tid < off) red[tid] += red[tid + off];
        __syncthreads();
    }
    if (tid == 0) sq[row] = red[0];
}

// -------------------- 2) W [D x OUT] f32 -> Wt [OUT x D] bf16 ---------------
__global__ __launch_bounds__(256)
void w_transpose_bf16(const float* __restrict__ W, uint16_t* __restrict__ Wt) {
    size_t idx = (size_t)blockIdx.x * 256 + threadIdx.x;  // over OUT*D
    int o = (int)(idx >> 12);          // / 4096
    int d = (int)(idx & 4095);         // % 4096
    Wt[idx] = f32_to_bf16(W[(size_t)d * OUTC + o]);
}

// -------------------- 3/4) 128x128-tile bf16 WMMA GEMM ----------------------
// Computes A @ B^T over row-major bf16 A,B (both K-stride = Kd).
// RBF=true : C -> exp(-0.5*(sq_i+sq_j-2*acc)) stored bf16 (ldC = 4096)
// RBF=false: C -> acc stored f32 (ldC = 1024)
template<bool RBF>
__global__ __launch_bounds__(256)
void gemm_bf16_128x128(const uint16_t* __restrict__ A,
                       const uint16_t* __restrict__ B,
                       void* __restrict__ Cout,
                       const float* __restrict__ sq,
                       int Kd, int ldC) {
    // double-buffered panels: 2 x (128 rows x 32 bf16, padded stride)
    __shared__ __align__(16) uint16_t AT[2][128 * PK];
    __shared__ __align__(16) uint16_t BT[2][128 * PK];

    const int tid  = threadIdx.x;
    const int lane = tid & 31;
    const int wid  = tid >> 5;          // 8 waves
    const int wm   = wid >> 1;          // 0..3 : 32-row band
    const int wn   = wid & 1;           // 0..1 : 64-col band
    const int i0   = blockIdx.y * 128;
    const int j0   = blockIdx.x * 128;

    const int fr   = lane & 15;         // fragment row/col within 16
    const int base = (lane >> 4) << 3;  // K-chunk base: 0 or 8 (ISA layout)

    // staging assignment: thread -> rows (r0, r0+64), 16B segment s0
    const int r0 = tid >> 2;            // 0..63
    const int s0 = tid & 3;             // 0..3

    auto gA = [&](int r, int k0) {
        return (const uint4*)(A + (size_t)(i0 + r) * Kd + k0) + s0;
    };
    auto gB = [&](int r, int k0) {
        return (const uint4*)(B + (size_t)(j0 + r) * Kd + k0) + s0;
    };

    v8f acc[2][4] = {};                 // 32x64 per wave

    // ---- prologue: stage tile 0 into buffer 0 ----
    {
        uint4 a0 = *gA(r0, 0), a1 = *gA(r0 + 64, 0);
        uint4 b0 = *gB(r0, 0), b1 = *gB(r0 + 64, 0);
        *(uint4*)&AT[0][r0 * PK + s0 * 8]        = a0;
        *(uint4*)&AT[0][(r0 + 64) * PK + s0 * 8] = a1;
        *(uint4*)&BT[0][r0 * PK + s0 * 8]        = b0;
        *(uint4*)&BT[0][(r0 + 64) * PK + s0 * 8] = b1;
    }
    __syncthreads();

    const int nk = Kd >> 5;             // k-steps of 32
    for (int ks = 0; ks < nk; ++ks) {
        const int cur  = ks & 1;
        const int nxt  = cur ^ 1;
        const bool more = (ks + 1) < nk;

        // issue next-tile global loads early (hide behind WMMAs)
        uint4 a0, a1, b0, b1;
        if (more) {
            const int k0 = (ks + 1) << 5;
            a0 = *gA(r0, k0);  a1 = *gA(r0 + 64, k0);
            b0 = *gB(r0, k0);  b1 = *gB(r0 + 64, k0);
        }

        // fragment loads from current buffer: two 16B chunks per lane
        FragBF aF[2], bF[4];
        #pragma unroll
        for (int mi = 0; mi < 2; ++mi) {
            int row = wm * 32 + mi * 16 + fr;
            aF[mi].u[0] = *(const uint4*)&AT[cur][row * PK + base];
            aF[mi].u[1] = *(const uint4*)&AT[cur][row * PK + base + 16];
        }
        #pragma unroll
        for (int ni = 0; ni < 4; ++ni) {
            int row = wn * 64 + ni * 16 + fr;
            bF[ni].u[0] = *(const uint4*)&BT[cur][row * PK + base];
            bF[ni].u[1] = *(const uint4*)&BT[cur][row * PK + base + 16];
        }

        #pragma unroll
        for (int mi = 0; mi < 2; ++mi)
            #pragma unroll
            for (int ni = 0; ni < 4; ++ni)
                acc[mi][ni] = __builtin_amdgcn_wmma_f32_16x16x32_bf16(
                    false, aF[mi].v, false, bF[ni].v,
                    (short)0, acc[mi][ni], false, false);

        // stage next tile into the other buffer (no conflict with cur reads)
        if (more) {
            *(uint4*)&AT[nxt][r0 * PK + s0 * 8]        = a0;
            *(uint4*)&AT[nxt][(r0 + 64) * PK + s0 * 8] = a1;
            *(uint4*)&BT[nxt][r0 * PK + s0 * 8]        = b0;
            *(uint4*)&BT[nxt][(r0 + 64) * PK + s0 * 8] = b1;
        }
        __syncthreads();   // single barrier per k-step
    }

    // epilogue: C layout -> lane<16: (M=r, N=lane); lane>=16: (M=r+8, N=lane-16)
    const int moff = (lane >> 4) << 3;
    #pragma unroll
    for (int mi = 0; mi < 2; ++mi) {
        #pragma unroll
        for (int ni = 0; ni < 4; ++ni) {
            #pragma unroll
            for (int r = 0; r < 8; ++r) {
                int gm = i0 + wm * 32 + mi * 16 + r + moff;
                int gn = j0 + wn * 64 + ni * 16 + fr;
                float v = acc[mi][ni][r];
                if constexpr (RBF) {
                    float kv = __expf(-0.5f * (sq[gm] + sq[gn] - 2.0f * v));
                    ((uint16_t*)Cout)[(size_t)gm * ldC + gn] = f32_to_bf16(kv);
                } else {
                    ((float*)Cout)[(size_t)gm * ldC + gn] = v;
                }
            }
        }
    }
}

// ---------------------------------------------------------------------------
extern "C" void kernel_launch(void* const* d_in, const int* in_sizes, int n_in,
                              void* d_out, int out_size, void* d_ws, size_t ws_size,
                              hipStream_t stream) {
    (void)in_sizes; (void)n_in; (void)out_size; (void)ws_size;

    const float* x = (const float*)d_in[0];  // [4096 x 4096]
    const float* W = (const float*)d_in[1];  // [4096 x 1024]
    float* out = (float*)d_out;              // [4096 x 1024]

    char* ws = (char*)d_ws;
    uint16_t* xb = (uint16_t*)(ws);                                  // 32 MB
    uint16_t* Kb = (uint16_t*)(ws + (size_t)32 * 1024 * 1024);       // 32 MB
    uint16_t* Wt = (uint16_t*)(ws + (size_t)64 * 1024 * 1024);       //  8 MB
    float*    sq = (float*)   (ws + (size_t)72 * 1024 * 1024);       // 16 KB

    // 1) bf16 conversion + row norms (norms from rounded data)
    rows_bf16_sqnorm<<<Dd, 256, 0, stream>>>(x, xb, sq);

    // 2) W -> Wt bf16 [OUT x D]
    w_transpose_bf16<<<(OUTC * Dd) / 256, 256, 0, stream>>>(W, Wt);

    // 3) Kb = exp(-0.5*(sq_i + sq_j - 2 * xb @ xb^T))  [4096 x 4096] bf16
    gemm_bf16_128x128<true><<<dim3(Dd / 128, Dd / 128), 256, 0, stream>>>(
        xb, xb, (void*)Kb, sq, Dd, Dd);

    // 4) out = Kb @ Wt^T  [4096 x 1024] f32
    gemm_bf16_128x128<false><<<dim3(OUTC / 128, Dd / 128), 256, 0, stream>>>(
        Kb, Wt, (void*)out, nullptr, Dd, OUTC);
}
